// CustomAttentionLayer_47914655154431
// MI455X (gfx1250) — matrix operand order repaired and checked by reading, not compiled, not run on hardware
//
#include <hip/hip_runtime.h>
#include <hip/hip_bf16.h>

typedef __bf16 bf16_t;
typedef __attribute__((ext_vector_type(16))) __bf16 v16bf;
typedef __attribute__((ext_vector_type(4)))  __bf16 v4bf;
typedef __attribute__((ext_vector_type(8)))  float  v8f;

#define H_DIM  2048
#define NH     16
#define NKV    4
#define HD     128
#define B_SZ   2
#define S_LEN  2048
#define M_ROWS (B_SZ * S_LEN)
#define HKV    (NKV * HD)
#define SCALE  0.08838834764831845f   // 1/sqrt(128)

union FragU { v16bf v; uint4 u[2]; };

__device__ __forceinline__ v8f vzero8() {
  v8f z;
  #pragma unroll
  for (int i = 0; i < 8; i++) z[i] = 0.0f;
  return z;
}

__device__ __forceinline__ v8f wmma_bf16(v16bf a, v16bf b, v8f c) {
  // D = A(16x32 bf16) x B(32x16 bf16) + C(16x16 f32)
  return __builtin_amdgcn_wmma_f32_16x16x32_bf16(
      false, a, false, b, (short)0, c, false, false);
}

// LDS byte offset of a __shared__ object: flat LDS aperture maps addr[31:0]
// to the wave's LDS offset (ISA 10.2 aperture table).
__device__ __forceinline__ uint32_t lds_off_u32(const void* p) {
  return (uint32_t)(uintptr_t)p;
}

// CDNA5 async global->LDS copy (no VGPR staging; tracked by ASYNCcnt).
__device__ __forceinline__ void async_copy_b128(uint32_t lds_addr, const void* g) {
  asm volatile("global_load_async_to_lds_b128 %0, %1, off"
               :: "v"(lds_addr), "v"((unsigned long long)(uintptr_t)g)
               : "memory");
}
__device__ __forceinline__ void wait_asynccnt0() {
  asm volatile("s_wait_asynccnt 0x0" ::: "memory");
}

// ---------------------------------------------------------------------------
// fp32 -> bf16 conversion (bandwidth pass, 4-wide)
// ---------------------------------------------------------------------------
__global__ void cvt_f32_to_bf16(const float4* __restrict__ x,
                                v4bf* __restrict__ y, int n4) {
  int i = blockIdx.x * blockDim.x + threadIdx.x;
  if (i >= n4) return;
  float4 f = x[i];
  v4bf o;
  o[0] = (__bf16)f.x; o[1] = (__bf16)f.y;
  o[2] = (__bf16)f.z; o[3] = (__bf16)f.w;
  y[i] = o;
}

// ---------------------------------------------------------------------------
// GEMM (NT): C[M,N] f32 = A[M,K] bf16 @ B[N,K]^T bf16
// Block tile 128x128, BK=64 (two WMMA K-steps per stage), 8 waves of 32x64.
// Software-pipelined with CDNA5 async global->LDS copies: the next stage
// streams into the ping-pong LDS buffer while WMMAs consume the current one;
// no VGPR staging, so nothing to spill.
// ---------------------------------------------------------------------------
#define BM 128
#define BN 128
#define BK 64
#define LDA 72   // padded LDS row stride in halves (144B, 16B-aligned)

__global__ __launch_bounds__(256) __attribute__((amdgpu_waves_per_eu(1)))
void gemm_bf16_nt(
    const bf16_t* __restrict__ A, const bf16_t* __restrict__ Bw,
    float* __restrict__ C, int M, int N, int K) {
  __shared__ bf16_t As[2][BM * LDA];
  __shared__ bf16_t Bs[2][BN * LDA];

  const int tid  = threadIdx.x;
  const int lane = tid & 31;
  const int wid  = tid >> 5;      // 0..7
  const int wm   = wid >> 1;      // 0..3 : 32 rows each
  const int wn   = wid & 1;       // 0..1 : 64 cols each
  const int half = lane >> 4;     // 0/1
  const int l15  = lane & 15;

  const int bm = blockIdx.y * BM;
  const int bn = blockIdx.x * BN;

  v8f acc[2][4];
  #pragma unroll
  for (int i = 0; i < 2; i++)
    #pragma unroll
    for (int j = 0; j < 4; j++) acc[i][j] = vzero8();

  // cooperative global->LDS mapping: each thread moves 32 halves of A and B
  const int lrow = tid >> 1;          // 0..127
  const int lk   = (tid & 1) * 32;    // 0 or 32

  const bf16_t* ag = A  + (size_t)(bm + lrow) * K + lk;
  const bf16_t* bg = Bw + (size_t)(bn + lrow) * K + lk;

  const uint32_t aoff[2] = { lds_off_u32(&As[0][lrow * LDA + lk]),
                             lds_off_u32(&As[1][lrow * LDA + lk]) };
  const uint32_t boff[2] = { lds_off_u32(&Bs[0][lrow * LDA + lk]),
                             lds_off_u32(&Bs[1][lrow * LDA + lk]) };

  // prologue: stage 0 streams straight into LDS buf 0
  #pragma unroll
  for (int c = 0; c < 4; c++) {
    async_copy_b128(aoff[0] + c * 16, ag + c * 8);
    async_copy_b128(boff[0] + c * 16, bg + c * 8);
  }
  wait_asynccnt0();
  __syncthreads();

  int buf = 0;
  for (int k0 = 0; k0 < K; k0 += BK) {
    const bool hasNext = (k0 + BK) < K;
    if (hasNext) {
      const int nb = buf ^ 1;
      #pragma unroll
      for (int c = 0; c < 4; c++) {
        async_copy_b128(aoff[nb] + c * 16, ag + k0 + BK + c * 8);
        async_copy_b128(boff[nb] + c * 16, bg + k0 + BK + c * 8);
      }
      if (k0 + 2 * BK < K) {
        __builtin_prefetch(ag + k0 + 2 * BK, 0, 1);   // global_prefetch_b8
        __builtin_prefetch(bg + k0 + 2 * BK, 0, 1);
      }
    }

    const bf16_t* as = As[buf];
    const bf16_t* bs = Bs[buf];
    #pragma unroll
    for (int kk = 0; kk < BK; kk += 32) {
      // A fragments: lanes 0-15 row M=l15 hold K{ko..ko+7, 16+ko..}, ko=half*8
      FragU af[2];
      #pragma unroll
      for (int mt = 0; mt < 2; mt++) {
        int row = wm * 32 + mt * 16 + l15;
        int ko  = kk + half * 8;
        af[mt].u[0] = *(const uint4*)&as[row * LDA + ko];
        af[mt].u[1] = *(const uint4*)&as[row * LDA + ko + 16];
      }
      // B fragments: lane col = l15, K{kb..kb+7, kb+16..kb+23}, kb=half*8
      FragU bfr[4];
      #pragma unroll
      for (int nt = 0; nt < 4; nt++) {
        int col = wn * 64 + nt * 16 + l15;
        int kb  = kk + half * 8;
        bfr[nt].u[0] = *(const uint4*)&bs[col * LDA + kb];
        bfr[nt].u[1] = *(const uint4*)&bs[col * LDA + kb + 16];
      }
      #pragma unroll
      for (int mt = 0; mt < 2; mt++)
        #pragma unroll
        for (int nt = 0; nt < 4; nt++)
          acc[mt][nt] = wmma_bf16(af[mt].v, bfr[nt].v, acc[mt][nt]);
    }

    if (hasNext) {
      wait_asynccnt0();       // own wave's async copies landed in LDS
      __syncthreads();        // all waves' copies landed
      buf ^= 1;
    }
  }

  // C layout: VGPR r -> row half*8 + r, col = l15
  #pragma unroll
  for (int mt = 0; mt < 2; mt++)
    #pragma unroll
    for (int nt = 0; nt < 4; nt++) {
      int col = bn + wn * 64 + nt * 16 + l15;
      #pragma unroll
      for (int r = 0; r < 8; r++) {
        int row = bm + wm * 32 + mt * 16 + half * 8 + r;
        C[(size_t)row * N + col] = acc[mt][nt][r];
      }
    }
}

// ---------------------------------------------------------------------------
// RoPE (interleaved pairs) + relayout [B,S,nh*HD] f32 -> [B,nh,S,HD] bf16
// ---------------------------------------------------------------------------
__global__ void rope_relayout(const float* __restrict__ src,
                              const float* __restrict__ cs,
                              const float* __restrict__ sn,
                              bf16_t* __restrict__ dst,
                              int nh, int dorope, int total) {
  int idx = blockIdx.x * blockDim.x + threadIdx.x;
  if (idx >= total) return;
  int i = idx & (HD / 2 - 1);          // pair index 0..63
  int t = idx >> 6;
  int h = t % nh;  t /= nh;
  int s = t % S_LEN;
  int b = t / S_LEN;
  const float* p = src + ((size_t)(b * S_LEN + s) * nh + h) * HD + 2 * i;
  float xr = p[0], xi = p[1], orr, oi;
  if (dorope) {
    float c  = cs[s * (HD / 2) + i];
    float ss = sn[s * (HD / 2) + i];
    orr = xr * c - xi * ss;
    oi  = xr * ss + xi * c;
  } else {
    orr = xr; oi = xi;
  }
  bf16_t* q = dst + ((size_t)(b * nh + h) * S_LEN + s) * HD + 2 * i;
  q[0] = (bf16_t)orr;
  q[1] = (bf16_t)oi;
}

// ---------------------------------------------------------------------------
// Flash attention: grid (S/64, B*NH), 4 waves x 16 q-rows, K-tiles of 32 keys.
// K tile streamed via async global->LDS; V tile register-transposed into LDS.
// Q,K,V bf16 head-major [B,heads,S,HD]; output bf16 [B,S,H].
// ---------------------------------------------------------------------------
#define LKT 136   // K tile row stride (HD=128 + 8 pad) halves
#define LVT 40    // V^T row stride (32 keys + 8 pad) halves
#define LPS 40    // P stage row stride
#define NKT (S_LEN / 32)

__global__ __launch_bounds__(128) __attribute__((amdgpu_waves_per_eu(1)))
void attn_kernel(
    const bf16_t* __restrict__ Q, const bf16_t* __restrict__ Kh,
    const bf16_t* __restrict__ Vh, bf16_t* __restrict__ Ob) {
  __shared__ bf16_t Kt[2][32 * LKT];        // 32 keys x 128 d
  __shared__ bf16_t VtT[2][HD * LVT];       // transposed: [d][key]
  __shared__ bf16_t Ps[4 * 16 * LPS];       // per-wave P stage 16x32

  const int tid  = threadIdx.x;
  const int lane = tid & 31;
  const int wid  = tid >> 5;     // 0..3
  const int half = lane >> 4;
  const int l15  = lane & 15;

  const int bh = blockIdx.y;
  const int b  = bh / NH, h = bh % NH;
  const int hk = h / (NH / NKV);          // GQA mapping
  const int q0 = blockIdx.x * 64 + wid * 16;

  const bf16_t* qp = Q  + (size_t)(b * NH  + h)  * S_LEN * HD;
  const bf16_t* kp = Kh + (size_t)(b * NKV + hk) * S_LEN * HD;
  const bf16_t* vp = Vh + (size_t)(b * NKV + hk) * S_LEN * HD;

  // resident Q fragments: 4 K-steps of 32 over HD=128
  FragU qf[4];
  {
    const bf16_t* qr = qp + (size_t)(q0 + l15) * HD;
    #pragma unroll
    for (int d = 0; d < 4; d++) {
      int db = d * 32 + half * 16;
      qf[d].u[0] = *(const uint4*)(qr + db);
      qf[d].u[1] = *(const uint4*)(qr + db + 8);
    }
  }

  v8f o[8];
  #pragma unroll
  for (int t = 0; t < 8; t++) o[t] = vzero8();
  float mr[8], lr[8];
  #pragma unroll
  for (int r = 0; r < 8; r++) { mr[r] = -3.0e38f; lr[r] = 0.0f; }

  const int krow = tid >> 2;            // 0..31 key row
  const int kcol = (tid & 3) * 32;      // d chunk
  bf16_t* psw = &Ps[wid * 16 * LPS];

  const uint32_t ktoff[2] = { lds_off_u32(&Kt[0][krow * LKT + kcol]),
                              lds_off_u32(&Kt[1][krow * LKT + kcol]) };

  uint4 vc[4];
  // prologue: tile 0 (K async -> LDS buf0, V via register transpose)
  {
    const bf16_t* ksrc = kp + (size_t)krow * HD + kcol;
    const bf16_t* vsrc = vp + (size_t)krow * HD + kcol;
    #pragma unroll
    for (int c = 0; c < 4; c++)
      async_copy_b128(ktoff[0] + c * 16, ksrc + c * 8);
    #pragma unroll
    for (int c = 0; c < 4; c++)
      vc[c] = *(const uint4*)(vsrc + c * 8);
    #pragma unroll
    for (int c = 0; c < 4; c++) {
      union { uint4 u; bf16_t hh[8]; } tmp;
      tmp.u = vc[c];
      #pragma unroll
      for (int j = 0; j < 8; j++)
        VtT[0][(kcol + c * 8 + j) * LVT + krow] = tmp.hh[j];
    }
  }
  wait_asynccnt0();
  __syncthreads();

  int buf = 0;
  for (int kt = 0; kt < NKT; kt++) {
    const bool hasNext = (kt + 1) < NKT;
    if (hasNext) {
      const bf16_t* ksrc = kp + (size_t)((kt + 1) * 32 + krow) * HD + kcol;
      const bf16_t* vsrc = vp + (size_t)((kt + 1) * 32 + krow) * HD + kcol;
      #pragma unroll
      for (int c = 0; c < 4; c++)
        async_copy_b128(ktoff[buf ^ 1] + c * 16, ksrc + c * 8);
      #pragma unroll
      for (int c = 0; c < 4; c++)
        vc[c] = *(const uint4*)(vsrc + c * 8);
      if (kt + 2 < NKT) {
        __builtin_prefetch(ksrc + 32 * HD, 0, 1);
        __builtin_prefetch(vsrc + 32 * HD, 0, 1);
      }
    }

    // scores S = Q @ K^T : 2 col-tiles (keys 0-15 / 16-31), 4 K-steps
    const bf16_t* ktb = Kt[buf];
    const bf16_t* vtb = VtT[buf];
    v8f s0 = vzero8(), s1 = vzero8();
    #pragma unroll
    for (int d = 0; d < 4; d++) {
      FragU kb0, kb1;
      int db = d * 32 + half * 16;
      kb0.u[0] = *(const uint4*)&ktb[l15 * LKT + db];
      kb0.u[1] = *(const uint4*)&ktb[l15 * LKT + db + 8];
      kb1.u[0] = *(const uint4*)&ktb[(16 + l15) * LKT + db];
      kb1.u[1] = *(const uint4*)&ktb[(16 + l15) * LKT + db + 8];
      s0 = wmma_bf16(qf[d].v, kb0.v, s0);
      s1 = wmma_bf16(qf[d].v, kb1.v, s1);
    }

    // online softmax (rows live in accumulator elements; cols across 16 lanes)
    float mx[8], corr[8], rs[8];
    #pragma unroll
    for (int r = 0; r < 8; r++) {
      s0[r] *= SCALE; s1[r] *= SCALE;
      mx[r] = fmaxf(s0[r], s1[r]);
    }
    for (int off = 1; off < 16; off <<= 1)
      #pragma unroll
      for (int r = 0; r < 8; r++)
        mx[r] = fmaxf(mx[r], __shfl_xor(mx[r], off, 32));
    #pragma unroll
    for (int r = 0; r < 8; r++) {
      float mn = fmaxf(mr[r], mx[r]);
      corr[r] = __expf(mr[r] - mn);
      mr[r] = mn;
      s0[r] = __expf(s0[r] - mn);
      s1[r] = __expf(s1[r] - mn);
      rs[r] = s0[r] + s1[r];
    }
    for (int off = 1; off < 16; off <<= 1)
      #pragma unroll
      for (int r = 0; r < 8; r++)
        rs[r] += __shfl_xor(rs[r], off, 32);
    #pragma unroll
    for (int r = 0; r < 8; r++) lr[r] = lr[r] * corr[r] + rs[r];
    #pragma unroll
    for (int t = 0; t < 8; t++)
      #pragma unroll
      for (int r = 0; r < 8; r++) o[t][r] *= corr[r];

    // stage P (C-layout -> A-layout via LDS), bf16
    #pragma unroll
    for (int r = 0; r < 8; r++) {
      psw[(half * 8 + r) * LPS + l15]      = (bf16_t)s0[r];
      psw[(half * 8 + r) * LPS + 16 + l15] = (bf16_t)s1[r];
    }
    __syncthreads();   // cross-lane LDS round-trip

    FragU pf;
    pf.u[0] = *(const uint4*)&psw[l15 * LPS + half * 16];
    pf.u[1] = *(const uint4*)&psw[l15 * LPS + half * 16 + 8];

    // O += P @ V : 8 d-tiles of 16
    #pragma unroll
    for (int t = 0; t < 8; t++) {
      FragU bv;
      int d = t * 16 + l15;
      bv.u[0] = *(const uint4*)&vtb[d * LVT + half * 16];
      bv.u[1] = *(const uint4*)&vtb[d * LVT + half * 16 + 8];
      o[t] = wmma_bf16(pf.v, bv.v, o[t]);
    }

    if (hasNext) {
      const int nbuf = buf ^ 1;
      #pragma unroll
      for (int c = 0; c < 4; c++) {
        union { uint4 u; bf16_t hh[8]; } tmp;
        tmp.u = vc[c];
        #pragma unroll
        for (int j = 0; j < 8; j++)
          VtT[nbuf][(kcol + c * 8 + j) * LVT + krow] = tmp.hh[j];
      }
      wait_asynccnt0();
      __syncthreads();
      buf = nbuf;
    }
  }

  // normalize and store to [B,S,H] bf16 (heads re-merged)
  #pragma unroll
  for (int r = 0; r < 8; r++) {
    float inv = 1.0f / lr[r];
    int srow = q0 + half * 8 + r;
    bf16_t* op = Ob + ((size_t)b * S_LEN + srow) * H_DIM + h * HD + l15;
    #pragma unroll
    for (int t = 0; t < 8; t++)
      op[t * 16] = (bf16_t)(o[t][r] * inv);
  }
}

// ---------------------------------------------------------------------------
// launch
// ---------------------------------------------------------------------------
extern "C" void kernel_launch(void* const* d_in, const int* in_sizes, int n_in,
                              void* d_out, int out_size, void* d_ws, size_t ws_size,
                              hipStream_t stream) {
  (void)in_sizes; (void)n_in; (void)out_size; (void)ws_size;
  const float* hs = (const float*)d_in[0];
  const float* fc = (const float*)d_in[1];
  const float* fs = (const float*)d_in[2];
  const float* Wq = (const float*)d_in[3];
  const float* Wk = (const float*)d_in[4];
  const float* Wv = (const float*)d_in[5];
  const float* Wo = (const float*)d_in[6];
  float* out = (float*)d_out;

  char* w = (char*)d_ws;
  auto alloc = [&](size_t bytes) {
    char* p = w;
    w += (bytes + 255) & ~(size_t)255;
    return p;
  };

  const size_t M = M_ROWS;
  bf16_t* hsb = (bf16_t*)alloc(M * H_DIM * 2);
  bf16_t* wqb = (bf16_t*)alloc((size_t)H_DIM * H_DIM * 2);
  bf16_t* wkb = (bf16_t*)alloc((size_t)HKV * H_DIM * 2);
  bf16_t* wvb = (bf16_t*)alloc((size_t)HKV * H_DIM * 2);
  bf16_t* wob = (bf16_t*)alloc((size_t)H_DIM * H_DIM * 2);
  float*  qfb = (float*)alloc(M * H_DIM * 4);
  float*  kfb = (float*)alloc(M * HKV * 4);
  float*  vfb = (float*)alloc(M * HKV * 4);
  bf16_t* qb  = (bf16_t*)alloc(M * H_DIM * 2);
  bf16_t* kb  = (bf16_t*)alloc(M * HKV * 2);
  bf16_t* vb  = (bf16_t*)alloc(M * HKV * 2);
  bf16_t* aob = (bf16_t*)alloc(M * H_DIM * 2);

  auto cvt = [&](const float* x, bf16_t* y, size_t n) {
    int n4 = (int)(n / 4);
    cvt_f32_to_bf16<<<(n4 + 255) / 256, 256, 0, stream>>>(
        (const float4*)x, (v4bf*)y, n4);
  };
  cvt(hs, hsb, M * H_DIM);
  cvt(Wq, wqb, (size_t)H_DIM * H_DIM);
  cvt(Wk, wkb, (size_t)HKV * H_DIM);
  cvt(Wv, wvb, (size_t)HKV * H_DIM);
  cvt(Wo, wob, (size_t)H_DIM * H_DIM);

  dim3 blk(256);
  gemm_bf16_nt<<<dim3(H_DIM / BN, M / BM), blk, 0, stream>>>(hsb, wqb, qfb, (int)M, H_DIM, H_DIM);
  gemm_bf16_nt<<<dim3(HKV / BN,   M / BM), blk, 0, stream>>>(hsb, wkb, kfb, (int)M, HKV, H_DIM);
  gemm_bf16_nt<<<dim3(HKV / BN,   M / BM), blk, 0, stream>>>(hsb, wvb, vfb, (int)M, HKV, H_DIM);

  int tq = B_SZ * S_LEN * NH * (HD / 2);
  int tk = B_SZ * S_LEN * NKV * (HD / 2);
  rope_relayout<<<(tq + 255) / 256, 256, 0, stream>>>(qfb, fc, fs, qb, NH, 1, tq);
  rope_relayout<<<(tk + 255) / 256, 256, 0, stream>>>(kfb, fc, fs, kb, NKV, 1, tk);
  rope_relayout<<<(tk + 255) / 256, 256, 0, stream>>>(vfb, fc, fs, vb, NKV, 0, tk);

  attn_kernel<<<dim3(S_LEN / 64, B_SZ * NH), dim3(128), 0, stream>>>(qb, kb, vb, aob);

  gemm_bf16_nt<<<dim3(H_DIM / BN, M / BM), blk, 0, stream>>>(aob, wob, out, (int)M, H_DIM, H_DIM);
}
